// MoELayer_26233660244556
// MI455X (gfx1250) — matrix operand order repaired and checked by reading, not compiled, not run on hardware
//
#include <hip/hip_runtime.h>
#include <hip/hip_bf16.h>
#include <math.h>
#include <stdint.h>

#define D_MODEL   1024
#define HIDDEN    4096
#define N_EXPERTS 8
#define T_TOKENS  2048
#define M_TILE    16
#define HC        128   // hidden chunk per iteration
#define BLOCK     256   // 8 waves (wave32)
#define XCH       256   // columns of f32 per TDM staging pass

typedef __attribute__((ext_vector_type(16))) __bf16 v16bf;
typedef __attribute__((ext_vector_type(8)))  __bf16 v8bf;
typedef __attribute__((ext_vector_type(8)))  float  v8f;
typedef __attribute__((ext_vector_type(4)))  float  v4f;
typedef __attribute__((ext_vector_type(4)))  unsigned int uint4v;
typedef __attribute__((ext_vector_type(8)))  int int8v;
typedef __attribute__((ext_vector_type(4)))  int int4v;

__device__ __forceinline__ __bf16 f2bf(float f) {
    __hip_bfloat16 h = __float2bfloat16(f);
    union { __hip_bfloat16 h; __bf16 b; } u;
    u.h = h;
    return u.b;
}

// branch-free tanh-form GELU: one v_exp_f32 (TRANS, co-executes with WMMA)
__device__ __forceinline__ float gelu_fast(float x) {
    float u = 0.7978845608028654f * x * (1.0f + 0.044715f * x * x);
    float t = 1.0f - 2.0f / (1.0f + __expf(2.0f * u));   // tanh(u)
    return 0.5f * x * (1.0f + t);
}

// ---------------------------------------------------------------- kernel 0
__global__ void moe_zero_counts(int* counts) {
    if (threadIdx.x < N_EXPERTS) counts[threadIdx.x] = 0;
}

// ---------------------------------------------------------------- kernel 1: router
__global__ void moe_router(const float* __restrict__ x,
                           const float* __restrict__ Wr,
                           int* __restrict__ counts,
                           int* __restrict__ buckets,
                           float* __restrict__ wtop) {
    const int wave = threadIdx.x >> 5;
    const int lane = threadIdx.x & 31;
    const int t    = blockIdx.x * (BLOCK / 32) + wave;
    if (t >= T_TOKENS) return;

    const float* xr = x + (size_t)t * D_MODEL;
    float acc[N_EXPERTS];
#pragma unroll
    for (int e = 0; e < N_EXPERTS; ++e) acc[e] = 0.0f;

    for (int d = lane; d < D_MODEL; d += 32) {
        const float xv = xr[d];
        const float* wr = Wr + (size_t)d * N_EXPERTS;
#pragma unroll
        for (int e = 0; e < N_EXPERTS; ++e) acc[e] += xv * wr[e];
    }
#pragma unroll
    for (int off = 16; off > 0; off >>= 1) {
#pragma unroll
        for (int e = 0; e < N_EXPERTS; ++e)
            acc[e] += __shfl_down(acc[e], off, 32);
    }
    if (lane == 0) {
        float l1 = -INFINITY, l2 = -INFINITY;
        int i1 = 0;
#pragma unroll
        for (int e = 0; e < N_EXPERTS; ++e) {
            float v = acc[e];
            if (v > l1) { l2 = l1; l1 = v; i1 = e; }
            else if (v > l2) { l2 = v; }
        }
        wtop[t] = 1.0f / (1.0f + __expf(l2 - l1));
        int pos = atomicAdd(&counts[i1], 1);
        buckets[i1 * T_TOKENS + pos] = t;
    }
}

// ---------------------------------------------------------------- kernel 2: fused expert FFN
struct __align__(16) MoeSmem {
    __bf16 X[M_TILE][D_MODEL];          // 32 KB: bf16 token tile
    union {
        float  xf[M_TILE][XCH];         // 16 KB: TDM gather landing pad (dead after staging)
        __bf16 H[M_TILE][HC];           //  4 KB: gelu(h) chunk
    } u;
    int   tok[M_TILE];
    float twt[M_TILE];
};

__global__ __launch_bounds__(BLOCK, 2)
void moe_ffn(const float* __restrict__ x,
             const float* __restrict__ W1,
             const float* __restrict__ W2,
             const int* __restrict__ counts,
             const int* __restrict__ buckets,
             const float* __restrict__ wtop,
             float* __restrict__ out) {
    __shared__ MoeSmem sm;

    const int e   = blockIdx.y;
    const int cnt = counts[e];
    const int t0  = blockIdx.x * M_TILE;
    if (t0 >= cnt) return;
    const int rows = min(M_TILE, cnt - t0);

    const int tid   = threadIdx.x;
    const int wave  = tid >> 5;
    const int lane  = tid & 31;
    const int col   = lane & 15;
    const int khalf = lane >> 4;

    if (tid < M_TILE) {
        const int r = (tid < rows) ? tid : 0;
        const int t = buckets[e * T_TOKENS + t0 + r];
        sm.tok[tid] = t;
        sm.twt[tid] = wtop[t];
    }

    // ---- build the TDM gather descriptor (all SGPR-uniform)
    // group2/3: sixteen 16-bit row indices = routed token ids (pad = last valid row)
    unsigned int ipack[8];
#pragma unroll
    for (int i = 0; i < 8; ++i) {
        const int r0 = min(2 * i,     rows - 1);
        const int r1 = min(2 * i + 1, rows - 1);
        const unsigned int ta = (unsigned int)
            __builtin_amdgcn_readfirstlane(buckets[e * T_TOKENS + t0 + r0]);
        const unsigned int tb = (unsigned int)
            __builtin_amdgcn_readfirstlane(buckets[e * T_TOKENS + t0 + r1]);
        ipack[i] = (ta & 0xFFFFu) | (tb << 16);
    }
    int4v g2, g3;
#pragma unroll
    for (int i = 0; i < 4; ++i) { g2[i] = (int)ipack[i]; g3[i] = (int)ipack[4 + i]; }

    // group1: data_size=4B, tensor_dim0=1024, tensor_dim1=2048 tokens,
    //         tile_dim0=XCH, tile_dim1=16 valid indices, dim0_stride=1024
    int8v g1;
    g1[0] = 0x20000;            // data_size = 2 (4 bytes); no multicast/pad/iterate
    g1[1] = (D_MODEL << 16);    // tensor_dim0[15:0] in d1[31:16]
    g1[2] = (T_TOKENS << 16);   // tensor_dim0[31:16]=0 | tensor_dim1[15:0]
    g1[3] = (XCH << 16);        // tensor_dim1[31:16]=0 | tile_dim0
    g1[4] = M_TILE;             // tile_dim1 = #valid gather indices; tile_dim2=0
    g1[5] = D_MODEL;            // tensor_dim0_stride[31:0]
    g1[6] = 0;
    g1[7] = 0;

    int8v gz;                   // zero-filled trailing group (6-arg builtin form)
#pragma unroll
    for (int i = 0; i < 8; ++i) gz[i] = 0;

    const unsigned int ldsOff = (unsigned int)(uintptr_t)&sm.u.xf[0][0];

    // ---- stage X tile via TDM gather, 4 passes of 16 x 256 f32, cvt -> bf16
    for (int p = 0; p < D_MODEL / XCH; ++p) {
        if (wave == 0) {
            const unsigned long long ga =
                (unsigned long long)(uintptr_t)x + (unsigned long long)p * (XCH * 4);
            uint4v g0;
            g0[0] = 0x80000001u;                 // count=1 | gather_mode | 16-bit idx
            g0[1] = ldsOff;                      // lds_addr
            g0[2] = (unsigned int)ga;            // global_addr[31:0]
            g0[3] = (unsigned int)(ga >> 32) | 0x80000000u;  // addr[56:32] | type=2
            __builtin_amdgcn_tensor_load_to_lds(g0, g1, g2, g3, gz, 0);
            __builtin_amdgcn_s_wait_tensorcnt(0);
        }
        __syncthreads();
        {
            const int r  = tid >> 4;                 // 16 threads per row
            const int c0 = (tid & 15) * (XCH / 16);  // 16 contiguous floats each
#pragma unroll
            for (int j = 0; j < XCH / 16; j += 4) {
                v4f v = *(const v4f*)&sm.u.xf[r][c0 + j];
                sm.X[r][p * XCH + c0 + j + 0] = f2bf(v.x);
                sm.X[r][p * XCH + c0 + j + 1] = f2bf(v.y);
                sm.X[r][p * XCH + c0 + j + 2] = f2bf(v.z);
                sm.X[r][p * XCH + c0 + j + 3] = f2bf(v.w);
            }
        }
        __syncthreads();
    }

    const float* W1e = W1 + (size_t)e * D_MODEL * HIDDEN;
    const float* W2e = W2 + (size_t)e * HIDDEN * D_MODEL;

    v8f yacc[8];
#pragma unroll
    for (int i = 0; i < 8; ++i)
#pragma unroll
        for (int j = 0; j < 8; ++j) yacc[i][j] = 0.0f;

    for (int hcb = 0; hcb < HIDDEN; hcb += HC) {
        // warm L2/L0 for next chunk's W1 columns while we compute this one
        if (hcb + HC < HIDDEN)
            __builtin_prefetch(W1e + (size_t)(khalf * 16) * HIDDEN
                               + hcb + HC + wave * 16 + col, 0, 1);

        // ======== GEMM1: h(16 x HC) = Xtile(16x1024) @ W1e[:, hcb:hcb+HC)
        const int hn0 = hcb + wave * 16;
        v8f hacc = {0.f, 0.f, 0.f, 0.f, 0.f, 0.f, 0.f, 0.f};
        for (int kk = 0; kk < D_MODEL; kk += 32) {
            const int arow  = col;
            const int kbase = kk + khalf * 16;
            v8bf alo = *(const v8bf*)&sm.X[arow][kbase];
            v8bf ahi = *(const v8bf*)&sm.X[arow][kbase + 8];
            v16bf a;
#pragma unroll
            for (int j = 0; j < 8; ++j) { a[j] = alo[j]; a[j + 8] = ahi[j]; }
            const int kb = kk + khalf * 16;
            const float* bp = W1e + (size_t)kb * HIDDEN + hn0 + col;
            v16bf b;
#pragma unroll
            for (int j = 0; j < 16; ++j) b[j] = f2bf(bp[(size_t)j * HIDDEN]);
            hacc = __builtin_amdgcn_wmma_f32_16x16x32_bf16(
                false, a, false, b, (short)0, hacc, false, false);
        }
        __syncthreads();   // previous chunk's H fully consumed
#pragma unroll
        for (int vg = 0; vg < 8; ++vg) {
            const int row = vg + khalf * 8;          // D-matrix row layout
            sm.u.H[row][wave * 16 + col] = f2bf(gelu_fast(hacc[vg]));
        }
        __syncthreads();

        // ======== GEMM2: yacc(16 x 128 / wave) += gelu_h(16 x HC) @ W2e[chunk, :]
        for (int kk = 0; kk < HC; kk += 32) {
            const int arow  = col;
            const int kbase = kk + khalf * 16;
            v8bf alo = *(const v8bf*)&sm.u.H[arow][kbase];
            v8bf ahi = *(const v8bf*)&sm.u.H[arow][kbase + 8];
            v16bf a;
#pragma unroll
            for (int j = 0; j < 8; ++j) { a[j] = alo[j]; a[j + 8] = ahi[j]; }
            const int kb = hcb + kk + khalf * 16;
#pragma unroll
            for (int nt = 0; nt < 8; ++nt) {
                const int n0 = wave * 128 + nt * 16;
                const float* bp = W2e + (size_t)kb * D_MODEL + n0 + col;
                v16bf b;
#pragma unroll
                for (int j = 0; j < 16; ++j) b[j] = f2bf(bp[(size_t)j * D_MODEL]);
                yacc[nt] = __builtin_amdgcn_wmma_f32_16x16x32_bf16(
                    false, a, false, b, (short)0, yacc[nt], false, false);
            }
        }
    }

    // ---- epilogue: scale by routing weight, scatter rows (fp32)
#pragma unroll
    for (int nt = 0; nt < 8; ++nt) {
        const int n0 = wave * 128 + nt * 16 + col;
#pragma unroll
        for (int vg = 0; vg < 8; ++vg) {
            const int r = vg + khalf * 8;
            if (r < rows) {
                out[(size_t)sm.tok[r] * D_MODEL + n0] = sm.twt[r] * yacc[nt][vg];
            }
        }
    }
}

// ---------------------------------------------------------------- launch
extern "C" void kernel_launch(void* const* d_in, const int* in_sizes, int n_in,
                              void* d_out, int out_size, void* d_ws, size_t ws_size,
                              hipStream_t stream) {
    const float* x  = (const float*)d_in[0];
    const float* Wr = (const float*)d_in[1];
    const float* W1 = (const float*)d_in[2];
    const float* W2 = (const float*)d_in[3];
    float* out = (float*)d_out;

    int*   counts  = (int*)d_ws;
    int*   buckets = counts + N_EXPERTS;
    float* wtop    = (float*)(buckets + N_EXPERTS * T_TOKENS);

    moe_zero_counts<<<1, 64, 0, stream>>>(counts);
    moe_router<<<T_TOKENS / (BLOCK / 32), BLOCK, 0, stream>>>(
        x, Wr, counts, buckets, wtop);

    dim3 grid(T_TOKENS / M_TILE, N_EXPERTS);
    moe_ffn<<<grid, BLOCK, 0, stream>>>(x, W1, W2, counts, buckets, wtop, out);
}
